// AutoregressiveSelfAttention_42021960024539
// MI455X (gfx1250) — matrix operand order repaired and checked by reading, not compiled, not run on hardware
//
#include <hip/hip_runtime.h>
#include <hip/hip_bf16.h>

// ---------------------------------------------------------------------------
// CDNA5 (gfx1250) bf16 WMMA attention block. All matmuls use
// v_wmma_f32_16x16x32_bf16 (fp32 accum); fragments are staged in LDS in
// per-lane fragment order so each fragment load is 1-2 ds_load_b128.
// ---------------------------------------------------------------------------

typedef __bf16 bf16t;
typedef __attribute__((ext_vector_type(16))) __bf16 v16bf;
typedef __attribute__((ext_vector_type(8)))  __bf16 v8bf;
typedef __attribute__((ext_vector_type(8)))  float  v8f;

#define WMMA_BF16(A, B, C) \
    __builtin_amdgcn_wmma_f32_16x16x32_bf16(false, (A), false, (B), (short)0, (C), false, false)

#define N_HEAD 12
#define L_SEQ  2048
#define BATCH  2
#define D_KQ   16
#define D_V    96
#define LD_KV  1344   // kv row stride (192 k cols + 1152 v cols)
#define LD_Q   192
#define LD_Y   1152
#define M_ROWS 4096   // B*L

// -------------------------- conversion kernels -----------------------------

__global__ void f32_to_bf16_kernel(const float* __restrict__ src,
                                   bf16t* __restrict__ dst, int n) {
    int i = blockIdx.x * blockDim.x + threadIdx.x;
    if (i < n) dst[i] = (bf16t)src[i];
}

// cat = concat(x, side) along feature dim, flattened to [4096, 1152] bf16
__global__ void build_cat_kernel(const float* __restrict__ x,
                                 const float* __restrict__ side,
                                 bf16t* __restrict__ cat) {
    int idx = blockIdx.x * blockDim.x + threadIdx.x;
    if (idx >= M_ROWS * 1152) return;
    int row = idx / 1152;
    int col = idx - row * 1152;
    float v;
    if (col < 768) v = x[(size_t)row * 768 + col];
    else           v = side[(size_t)row * 384 + (col - 768)];
    cat[idx] = (bf16t)v;
}

// ------------------------------ GEMM kernel --------------------------------
// C[M,N] = A[M,K] @ B[K,N] + bias; bf16 in, fp32 accum.
// Block = 128 threads (4 waves). Tile 128x64, K stepped by 32.
// Wave w computes rows [m0 + 32w, +32) (two 16-row sub-tiles) x 64 cols.
// A rows stored in LDS K-permuted [0..7|16..23|8..15|24..31] so each lane
// group's 16-half A fragment is contiguous. B stored in fragment order
// BsF[ntile][lane][k] so each B fragment is a contiguous 32-byte load.

__global__ __launch_bounds__(128)
void gemm_bf16_kernel(const bf16t* __restrict__ A, int lda,
                      const bf16t* __restrict__ B, int ldb,
                      const float* __restrict__ bias,
                      bf16t* __restrict__ Cb, float* __restrict__ Cf, int ldc,
                      int K) {
    __shared__ __align__(16) bf16t AsP[128][40];      // 128x32, permuted K
    __shared__ __align__(16) bf16t BsF[4][32][16];    // fragment layout

    const int tid  = threadIdx.x;
    const int wave = tid >> 5;
    const int lane = tid & 31;
    const int g    = lane >> 4;
    const int ln   = lane & 15;
    const int m0   = blockIdx.y * 128;
    const int n0   = blockIdx.x * 64;

    v8f acc[2][4] = {};

    for (int k0 = 0; k0 < K; k0 += 32) {
        // --- stage A: 128x32 halves as 512 8-wide chunks, 4 per thread ---
        #pragma unroll
        for (int it = 0; it < 4; ++it) {
            int ch = tid + it * 128;
            int r  = ch >> 2;
            int c8 = (ch & 3) * 8;                     // 0,8,16,24
            v8bf d = *(const v8bf*)&A[(size_t)(m0 + r) * lda + k0 + c8];
            int pos = (c8 == 8) ? 16 : (c8 == 16) ? 8 : c8;  // K permutation
            *(v8bf*)&AsP[r][pos] = d;
        }
        // --- stage B: 32x64 halves as 256 8-wide chunks, 2 per thread,
        //     scatter-stored into fragment layout ---
        #pragma unroll
        for (int it = 0; it < 2; ++it) {
            int ch = tid + it * 128;
            int r  = ch >> 3;                          // k row 0..31
            int c8 = (ch & 7) * 8;                     // col 0..56
            v8bf d = *(const v8bf*)&B[(size_t)(k0 + r) * ldb + n0 + c8];
            int gg = (r >> 3) & 1, hh = (r & 7) | ((r >> 4) << 3);
            #pragma unroll
            for (int e = 0; e < 8; ++e) {
                int c = c8 + e;
                BsF[c >> 4][gg * 16 + (c & 15)][hh] = d[e];
            }
        }
        __syncthreads();

        v16bf af0 = *(const v16bf*)&AsP[wave * 32 + ln][g * 16];
        v16bf af1 = *(const v16bf*)&AsP[wave * 32 + 16 + ln][g * 16];
        #pragma unroll
        for (int t = 0; t < 4; ++t) {
            v16bf bfr = *(const v16bf*)&BsF[t][lane][0];
            acc[0][t] = WMMA_BF16(af0, bfr, acc[0][t]);
            acc[1][t] = WMMA_BF16(af1, bfr, acc[1][t]);
        }
        __syncthreads();
    }

    // epilogue: row = m0 + wave*32 + s*16 + g*8 + r, col = n0 + t*16 + ln
    #pragma unroll
    for (int s = 0; s < 2; ++s)
        #pragma unroll
        for (int t = 0; t < 4; ++t) {
            int col = n0 + t * 16 + ln;
            float bv = bias ? bias[col] : 0.0f;
            #pragma unroll
            for (int r = 0; r < 8; ++r) {
                int row = m0 + wave * 32 + s * 16 + g * 8 + r;
                float v = acc[s][t][r] + bv;
                if (Cb) Cb[(size_t)row * ldc + col] = (bf16t)v;
                if (Cf) Cf[(size_t)row * ldc + col] = v;
            }
        }
}

// --------------------------- attention kernel ------------------------------
// Block = 4 waves; wave w handles query tile qt = blockIdx.x*4 + w.
// Outer loop over 32-key blocks shared by all 4 waves (V staged once in LDS,
// fragment order). Per block: 2 score WMMAs (dk=16 padded to K=32) and
// 6 full-K P@V WMMAs. Mask: j < i, except row 0 attends j == 0.

__global__ __launch_bounds__(128)
void attn_kernel(const bf16t* __restrict__ qb,
                 const bf16t* __restrict__ kvb,
                 bf16t* __restrict__ yb) {
    const int tid  = threadIdx.x;
    const int wave = tid >> 5;
    const int lane = tid & 31;
    const int g    = lane >> 4;
    const int n    = lane & 15;
    const int h    = blockIdx.y;
    const int b    = blockIdx.z;
    const int qt   = blockIdx.x * 4 + wave;
    const int i0   = qt * 16;

    const bf16t* q = qb  + (size_t)b * L_SEQ * LD_Q  + h * D_KQ;
    const bf16t* k = kvb + (size_t)b * L_SEQ * LD_KV + h * D_KQ;
    const bf16t* v = kvb + (size_t)b * L_SEQ * LD_KV + 192 + h * D_V;
    bf16t*       y = yb  + (size_t)b * L_SEQ * LD_Y  + h * D_V;

    __shared__ __align__(16) bf16t VlsF[6][32][16];   // V block, fragment order
    __shared__ __align__(16) bf16t Pls[4][16][32];    // per-wave P (row-major)

    const int kb8 = g * 8;

    // q A-fragment: halves 0..7 = q[i0+n][kb8..kb8+7], halves 8..15 = 0
    union { v16bf v16; v8bf v8[2]; } qu;
    qu.v8[0] = *(const v8bf*)&q[(size_t)(i0 + n) * LD_Q + kb8];
    { v8bf z = {}; qu.v8[1] = z; }
    const v16bf qf = qu.v16;

    float rm[8], rl[8];
    v8f o[6] = {};
    #pragma unroll
    for (int r = 0; r < 8; ++r) { rm[r] = -__builtin_inff(); rl[r] = 0.0f; }

    // key blocks of 32 covering keys 0 .. i0_max+15 for the whole block
    const int i0max = (blockIdx.x * 4 + 3) * 16;
    const int nbmax = (i0max + 15) / 32 + 1;

    for (int kb = 0; kb < nbmax; ++kb) {
        const int j0 = kb * 32;

        // --- stage V block 32x96 into fragment layout (whole workgroup) ---
        #pragma unroll
        for (int it = 0; it < 3; ++it) {
            int ch = tid + it * 128;                  // 384 chunks of 8
            int r  = ch / 12;                         // key row 0..31
            int c8 = (ch - r * 12) * 8;               // col 0..88
            v8bf d = *(const v8bf*)&v[(size_t)(j0 + r) * LD_KV + c8];
            int base = ((r >> 3) & 1) * 16, hh = (r & 7) | ((r >> 4) << 3);
            #pragma unroll
            for (int e = 0; e < 8; ++e) {
                int c = c8 + e;
                VlsF[c >> 4][base + (c & 15)][hh] = d[e];
            }
        }
        __syncthreads();

        // --- scores for keys j0..j0+31 (two 16-key B fragments) ---
        union { v16bf v16; v8bf v8[2]; } ku0, ku1;
        ku0.v8[0] = *(const v8bf*)&k[(size_t)(j0 + n) * LD_KV + kb8];
        ku1.v8[0] = *(const v8bf*)&k[(size_t)(j0 + 16 + n) * LD_KV + kb8];
        { v8bf z = {}; ku0.v8[1] = z; ku1.v8[1] = z; }
        const v16bf kf0 = ku0.v16, kf1 = ku1.v16;
        v8f s0 = {}, s1 = {};
        s0 = WMMA_BF16(qf, kf0, s0);
        s1 = WMMA_BF16(qf, kf1, s1);

        // --- masked online softmax over 32 keys ---
        float alpha[8];
        #pragma unroll
        for (int r = 0; r < 8; ++r) {
            int i  = i0 + g * 8 + r;
            int ja = j0 + n;
            int jb = j0 + 16 + n;
            float sv0 = s0[r] * 0.25f;               // 1/sqrt(16)
            float sv1 = s1[r] * 0.25f;
            if (!((ja < i) || (i == 0 && ja == 0))) sv0 = -__builtin_inff();
            if (!(jb < i))                          sv1 = -__builtin_inff();

            float x = fmaxf(sv0, sv1);
            #pragma unroll
            for (int off = 1; off < 16; off <<= 1)
                x = fmaxf(x, __shfl_xor(x, off, 32));
            float nm = fmaxf(rm[r], x);
            float a  = __expf(rm[r] - nm);
            float p0 = __expf(sv0 - nm);
            float p1 = __expf(sv1 - nm);
            float rs = p0 + p1;
            #pragma unroll
            for (int off = 1; off < 16; off <<= 1)
                rs += __shfl_xor(rs, off, 32);
            rl[r] = rl[r] * a + rs;
            rm[r] = nm;
            alpha[r] = a;
            Pls[wave][g * 8 + r][n]      = (bf16t)p0;
            Pls[wave][g * 8 + r][16 + n] = (bf16t)p1;
        }

        #pragma unroll
        for (int t = 0; t < 6; ++t)
            #pragma unroll
            for (int r = 0; r < 8; ++r) o[t][r] *= alpha[r];

        // --- P (16x32, full K) as A-fragment from per-wave LDS ---
        union { v16bf v16; v8bf v8[2]; } pu;
        pu.v8[0] = *(const v8bf*)&Pls[wave][n][kb8];
        pu.v8[1] = *(const v8bf*)&Pls[wave][n][16 + kb8];
        const v16bf pf = pu.v16;

        #pragma unroll
        for (int t = 0; t < 6; ++t) {
            v16bf vf = *(const v16bf*)&VlsF[t][lane][0];
            o[t] = WMMA_BF16(pf, vf, o[t]);
        }
        __syncthreads();   // VlsF fully consumed before next staging
    }

    // normalize and store (token 0 rows overwritten by first_token_kernel)
    #pragma unroll
    for (int t = 0; t < 6; ++t)
        #pragma unroll
        for (int r = 0; r < 8; ++r) {
            int row = i0 + g * 8 + r;
            float val = o[t][r] / rl[r];
            y[(size_t)row * LD_Y + t * 16 + n] = (bf16t)val;
        }
}

// ------------------- token-0 special case (fp32, tiny) ---------------------

__global__ void first_token_kernel(const float* __restrict__ side,
                                   const float* __restrict__ Wemb,
                                   const float* __restrict__ bemb,
                                   bf16t* __restrict__ yb) {
    int idx = blockIdx.x * blockDim.x + threadIdx.x;
    if (idx >= BATCH * 1152) return;
    int b = idx / 1152, nn = idx - b * 1152;
    const float* s = side + (size_t)b * L_SEQ * 384;   // side[b, 0, :]
    float acc = bemb[nn];
    for (int kk = 0; kk < 384; ++kk)
        acc += s[kk] * Wemb[(size_t)kk * 1152 + nn];
    yb[(size_t)b * L_SEQ * LD_Y + nn] = (bf16t)acc;
}

// ------------------------------- launcher ----------------------------------

extern "C" void kernel_launch(void* const* d_in, const int* in_sizes, int n_in,
                              void* d_out, int out_size, void* d_ws, size_t ws_size,
                              hipStream_t stream) {
    (void)in_sizes; (void)n_in; (void)out_size; (void)ws_size;

    const float* x     = (const float*)d_in[0];
    const float* side  = (const float*)d_in[1];
    const float* Wq    = (const float*)d_in[2];
    const float* bq    = (const float*)d_in[3];
    const float* Wkv   = (const float*)d_in[4];
    const float* bkv   = (const float*)d_in[5];
    const float* Wproj = (const float*)d_in[6];
    const float* bproj = (const float*)d_in[7];
    const float* Wemb  = (const float*)d_in[8];
    const float* bemb  = (const float*)d_in[9];
    float* out = (float*)d_out;

    char* ws = (char*)d_ws;
    size_t off = 0;
    auto salloc = [&](size_t bytes) -> void* {
        void* p = ws + off;
        off += (bytes + 255) & ~(size_t)255;
        return p;
    };
    bf16t* cat_b   = (bf16t*)salloc((size_t)M_ROWS * 1152 * 2);
    bf16t* q_b     = (bf16t*)salloc((size_t)M_ROWS * 192 * 2);
    bf16t* kv_b    = (bf16t*)salloc((size_t)M_ROWS * 1344 * 2);
    bf16t* y_b     = (bf16t*)salloc((size_t)M_ROWS * 1152 * 2);
    bf16t* Wq_b    = (bf16t*)salloc((size_t)384 * 192 * 2);
    bf16t* Wkv_b   = (bf16t*)salloc((size_t)1152 * 1344 * 2);
    bf16t* Wproj_b = (bf16t*)salloc((size_t)1152 * 1152 * 2);

    // 1) precision conversion
    {
        int n = M_ROWS * 1152;
        build_cat_kernel<<<(n + 255) / 256, 256, 0, stream>>>(x, side, cat_b);
    }
    {
        int n = 384 * 192;
        f32_to_bf16_kernel<<<(n + 255) / 256, 256, 0, stream>>>(Wq, Wq_b, n);
    }
    {
        int n = 1152 * 1344;
        f32_to_bf16_kernel<<<(n + 255) / 256, 256, 0, stream>>>(Wkv, Wkv_b, n);
    }
    {
        int n = 1152 * 1152;
        f32_to_bf16_kernel<<<(n + 255) / 256, 256, 0, stream>>>(Wproj, Wproj_b, n);
    }

    // 2) q = side @ Wq + bq            [4096,192], K=384 (side = cat cols 768..)
    gemm_bf16_kernel<<<dim3(192 / 64, M_ROWS / 128), 128, 0, stream>>>(
        cat_b + 768, 1152, Wq_b, 192, bq, q_b, nullptr, 192, 384);

    // 3) kv = cat @ Wkv + bkv          [4096,1344], K=1152
    gemm_bf16_kernel<<<dim3(1344 / 64, M_ROWS / 128), 128, 0, stream>>>(
        cat_b, 1152, Wkv_b, 1344, bkv, kv_b, nullptr, 1344, 1152);

    // 4) attention -> y_b bf16 [4096,1152]
    attn_kernel<<<dim3(L_SEQ / 16 / 4, N_HEAD, BATCH), 128, 0, stream>>>(
        q_b, kv_b, y_b);

    // 5) overwrite token-0 rows with side[:,0] @ Wemb + bemb
    first_token_kernel<<<(BATCH * 1152 + 255) / 256, 256, 0, stream>>>(
        side, Wemb, bemb, y_b);

    // 6) out = y @ Wproj + bproj       fp32 [4096,1152], K=1152
    gemm_bf16_kernel<<<dim3(1152 / 64, M_ROWS / 128), 128, 0, stream>>>(
        y_b, 1152, Wproj_b, 1152, bproj, nullptr, out, 1152, 1152);
}
